// SSMLayer_54606214201759
// MI455X (gfx1250) — matrix operand order repaired and checked by reading, not compiled, not run on hardware
//
#include <hip/hip_runtime.h>
#include <hip/hip_bf16.h>
#include <stdint.h>

#define B_ 8
#define T_ 2048
#define H_ 1024
#define S_ 512

typedef __attribute__((ext_vector_type(16))) __bf16       v16bf;
typedef __attribute__((ext_vector_type(8)))  float        v8f;
typedef __attribute__((ext_vector_type(4)))  unsigned int u32x4;
typedef __attribute__((ext_vector_type(8)))  int          i32x8;
typedef __attribute__((ext_vector_type(4)))  int          i32x4;

#if defined(__has_builtin)
#  if __has_builtin(__builtin_amdgcn_tensor_load_to_lds)
#    define HAVE_TDM 1
#  endif
#endif
#ifndef HAVE_TDM
#  define HAVE_TDM 0
#endif

static __device__ __forceinline__ int lane_id() { return (int)(threadIdx.x & 31u); }

// ---- WMMA fragment loaders (layouts per CDNA5 ISA §7.12.2) ----
// A-operand 16x32 (16-bit): lane l -> row = l&15, VGPR0-3 = K[(l>>4)*8 .. +7],
// VGPR4-7 = K[16+(l>>4)*8 .. +7]  => two contiguous 8-element runs.
static __device__ __forceinline__ v16bf frag_a_f32(const float* base, int ldr) {
  const int l = lane_id();
  const float* p0 = base + (size_t)(l & 15) * ldr + ((l >> 4) * 8);
  v16bf f;
#pragma unroll
  for (int i = 0; i < 8; ++i) { f[i] = (__bf16)p0[i]; f[8 + i] = (__bf16)p0[16 + i]; }
  return f;
}
static __device__ __forceinline__ v16bf frag_a_bf16(const __bf16* base, int ldr) {
  const int l = lane_id();
  const __bf16* p0 = base + (size_t)(l & 15) * ldr + ((l >> 4) * 8);
  v16bf f;
#pragma unroll
  for (int i = 0; i < 8; ++i) { f[i] = p0[i]; f[8 + i] = p0[16 + i]; }
  return f;
}
// B-operand 32x16 (16-bit): same K order as A: elems 0-7 = K[(l>>4)*8 .. +7], 8-15 = K[16+(l>>4)*8 .. +7].
// With B'[k][n] = W[n][k] (weights row-major) -> lane reads two 8-elem runs of row n.
static __device__ __forceinline__ v16bf frag_b_f32(const float* base, int ldr) {
  const int l = lane_id();
  const float* p = base + (size_t)(l & 15) * ldr + ((l >> 4) * 8);
  v16bf f;
#pragma unroll
  for (int i = 0; i < 8; ++i) { f[i] = (__bf16)p[i]; f[8 + i] = (__bf16)p[16 + i]; }
  return f;
}
static __device__ __forceinline__ v16bf frag_b_bf16(const __bf16* base, int ldr) {
  const int l = lane_id();
  const __bf16* p = base + (size_t)(l & 15) * ldr + ((l >> 4) * 8);
  v16bf f;
#pragma unroll
  for (int i = 0; i < 8; ++i) { f[i] = p[i]; f[8 + i] = p[16 + i]; }
  return f;
}
static __device__ __forceinline__ v8f wmma_bf16(v16bf a, v16bf b, v8f c) {
  return __builtin_amdgcn_wmma_f32_16x16x32_bf16(false, a, false, b, (short)0, c, false, false);
}

static __device__ __forceinline__ void wait_tensor() {
#if HAVE_TDM
  __builtin_amdgcn_s_wait_tensorcnt(0);
#endif
}

// TDM: 2D tile load x_state[b=0..7][t][n0..n0+31] (f32) -> LDS, D# per CDNA5 ISA
// This toolchain exposes the 6-arg builtin:
//   (uint32x4 g0, int32x8 g1, int32x4 g2, int32x4 g3, int32x8 pad, i32 cpol)
static __device__ __forceinline__ void tdm_load_xs(const float* xs, int t, int n0,
                                                   void* lds_dst) {
#if HAVE_TDM
  const uint64_t gaddr = (uint64_t)(uintptr_t)(xs + (size_t)t * S_ + n0);
  const unsigned ldsOff = (unsigned)(uintptr_t)lds_dst;  // LDS flat addr low 32 = offset
  u32x4 g0;
  g0[0] = 1u;                                            // count=1 (valid D#)
  g0[1] = ldsOff;                                        // lds_addr
  g0[2] = (unsigned)(gaddr & 0xffffffffu);               // global_addr[31:0]
  g0[3] = (unsigned)((gaddr >> 32) & 0x01ffffffu) | (2u << 30);  // addr[56:32] | type=2
  const unsigned td0 = 32u, td1 = 8u, tile0 = 32u, tile1 = 8u;
  const uint64_t s0 = (uint64_t)T_ * S_;                 // batch stride (data_size units)
  i32x8 g1;
  g1[0] = (int)(2u << 16);                               // data_size=2 (4 bytes)
  g1[1] = (int)((td0 & 0xffffu) << 16);                  // tensor_dim0[15:0]
  g1[2] = (int)((td0 >> 16) | ((td1 & 0xffffu) << 16));  // td0 hi | tensor_dim1 lo
  g1[3] = (int)((td1 >> 16) | (tile0 << 16));            // td1 hi | tile_dim0
  g1[4] = (int)(tile1);                                  // tile_dim1 | tile_dim2=0
  g1[5] = (int)(s0 & 0xffffffffu);                       // tensor_dim0_stride lo
  g1[6] = (int)((s0 >> 32) & 0xffffu);                   // stride hi | dim1_stride lo
  g1[7] = 0;
  const i32x4 z4 = {0, 0, 0, 0};
  const i32x8 z8 = {0, 0, 0, 0, 0, 0, 0, 0};
  __builtin_amdgcn_tensor_load_to_lds(g0, g1, z4, z4, z8, 0);
#else
  // fallback: cooperative plain load (all 256 threads)
  const int tid = (int)threadIdx.x;
  const int b = tid >> 5, j = tid & 31;
  ((float*)lds_dst)[b * 32 + j] = xs[((size_t)b * T_ + t) * S_ + n0 + j];
#endif
}

// ---------------- Kernel 1: x_state = x @ W_in^T + b_in ----------------
__global__ __launch_bounds__(256) void k_inproj(const float* __restrict__ x,
                                                const float* __restrict__ W_in,
                                                const float* __restrict__ b_in,
                                                float* __restrict__ xs) {
  const int m0 = blockIdx.x * 16;
  const int wave = (int)(threadIdx.x >> 5);
  const int l = lane_id();
#pragma unroll 1
  for (int j = 0; j < 4; ++j) {
    const int n0 = (wave * 4 + j) * 16;
    v8f acc = {};
#pragma unroll 1
    for (int k = 0; k < H_; k += 32) {
      v16bf a = frag_a_f32(x + (size_t)m0 * H_ + k, H_);
      v16bf b = frag_b_f32(W_in + (size_t)n0 * H_ + k, H_);
      acc = wmma_bf16(a, b, acc);
    }
    const int n = n0 + (l & 15);
    const int mb = (l >> 4) * 8;
    const float bias = b_in[n];
#pragma unroll
    for (int v = 0; v < 8; ++v)
      xs[(size_t)(m0 + mb + v) * S_ + n] = acc[v] + bias;
  }
}

// ---------------- Kernel 2: sequential scan ----------------
// 16 WGs; WG g owns output columns [g*32, g*32+32) with its A-row slice LDS-resident
// as bf16. Per step: h_new = clip(h @ A_slice^T + xs_t*Bv + bias_A). Cross-WG h
// exchange through hs[t] in L2 + monotonic atomic barrier. xs_t prefetched via TDM.
__global__ __launch_bounds__(256) void k_scan(const float* __restrict__ xs,
                                              const float* __restrict__ A,
                                              const float* __restrict__ Bv,
                                              const float* __restrict__ bias_A,
                                              float* __restrict__ hs,
                                              unsigned int* __restrict__ cnt) {
  __shared__ __bf16 Asl[32][520];   // A rows n0..n0+31, padded vs bank conflicts
  __shared__ __bf16 hb[16][520];    // h (batches 0-7 real, 8-15 zero padding)
  __shared__ float xsb[2][B_][32];  // double-buffered x_state tile
  __shared__ float bvs[32], bas[32];
  const int tid = (int)threadIdx.x;
  const int wave = tid >> 5;
  const int n0 = blockIdx.x * 32;

  {  // load A slice once (f32 -> bf16)
    const int r = tid >> 3, c0 = (tid & 7) * 64;
#pragma unroll 1
    for (int c = 0; c < 64; ++c)
      Asl[r][c0 + c] = (__bf16)A[(size_t)(n0 + r) * S_ + c0 + c];
  }
  for (int i = tid; i < 16 * 520; i += 256) (&hb[0][0])[i] = (__bf16)0.0f;
  if (tid < 32) { bvs[tid] = Bv[n0 + tid]; bas[tid] = bias_A[n0 + tid]; }

#if HAVE_TDM
  if (wave == 0) { tdm_load_xs(xs, 0, n0, &xsb[0][0][0]); wait_tensor(); }
#else
  tdm_load_xs(xs, 0, n0, &xsb[0][0][0]);
#endif
  __syncthreads();

  for (int t = 0; t < T_; ++t) {
    // prefetch xs[t+1] while computing step t
    if (t + 1 < T_) {
#if HAVE_TDM
      if (wave == 0) tdm_load_xs(xs, t + 1, n0, &xsb[(t + 1) & 1][0][0]);
#else
      tdm_load_xs(xs, t + 1, n0, &xsb[(t + 1) & 1][0][0]);
#endif
    }
    if (wave < 2) {
      v8f acc = {};
      const __bf16* bbase = &Asl[wave * 16][0];
#pragma unroll 1
      for (int k = 0; k < S_; k += 32) {
        v16bf a = frag_a_bf16(&hb[0][k], 520);
        v16bf b = frag_b_bf16(bbase + k, 520);
        acc = wmma_bf16(a, b, acc);
      }
      const int l = lane_id();
      if (l < 16) {  // lanes 0-15 hold batches 0-7 (M = v)
        const int nloc = wave * 16 + l;
        const int col = n0 + nloc;
        const float bvv = bvs[nloc], bav = bas[nloc];
#pragma unroll
        for (int v = 0; v < 8; ++v) {
          float val = acc[v] + xsb[t & 1][v][nloc] * bvv + bav;
          val = fminf(10.0f, fmaxf(-10.0f, val));
          __hip_atomic_store(&hs[((size_t)v * T_ + t) * S_ + col], val,
                             __ATOMIC_RELAXED, __HIP_MEMORY_SCOPE_AGENT);
        }
      }
    }
    __syncthreads();
    if (tid == 0) {
      __hip_atomic_fetch_add(cnt, 1u, __ATOMIC_RELEASE, __HIP_MEMORY_SCOPE_AGENT);
      const unsigned target = 16u * (unsigned)(t + 1);
      while (__hip_atomic_load(cnt, __ATOMIC_ACQUIRE, __HIP_MEMORY_SCOPE_AGENT) < target)
        __builtin_amdgcn_s_sleep(1);
    }
    __syncthreads();
    if (t + 1 < T_) {  // gather full h_t -> hb (bf16)
      const int b = tid >> 5;
      const int c0 = (tid & 31) * 16;
#pragma unroll
      for (int c = 0; c < 16; ++c) {
        float v = __hip_atomic_load(&hs[((size_t)b * T_ + t) * S_ + c0 + c],
                                    __ATOMIC_RELAXED, __HIP_MEMORY_SCOPE_AGENT);
        hb[b][c0 + c] = (__bf16)v;
      }
#if HAVE_TDM
      if (wave == 0) wait_tensor();
#endif
    }
    __syncthreads();
  }
}

// ------- Kernel 3: fused y = hs@C^T+bC; out = sig(x@Wg^T+bg)*(y@Wp^T+bp) + ... -------
__global__ __launch_bounds__(256) void k_out(const float* __restrict__ hs,
                                             const float* __restrict__ C,
                                             const float* __restrict__ bias_C,
                                             const float* __restrict__ x,
                                             const float* __restrict__ Wg,
                                             const float* __restrict__ bg,
                                             const float* __restrict__ Wp,
                                             const float* __restrict__ bp,
                                             float* __restrict__ out) {
  __shared__ __bf16 ytile[16][1032];  // y rows kept on-chip (bf16), padded
  const int r0 = blockIdx.x * 16;
  const int wave = (int)(threadIdx.x >> 5);
  const int l = lane_id();
  // Phase 1: y = hs_tile @ C^T + bias_C  -> ytile
#pragma unroll 1
  for (int j = 0; j < 8; ++j) {
    const int n0 = (wave * 8 + j) * 16;
    v8f acc = {};
#pragma unroll 1
    for (int k = 0; k < S_; k += 32) {
      v16bf a = frag_a_f32(hs + (size_t)r0 * S_ + k, S_);
      v16bf b = frag_b_f32(C + (size_t)n0 * S_ + k, S_);
      acc = wmma_bf16(a, b, acc);
    }
    const int n = n0 + (l & 15);
    const int mb = (l >> 4) * 8;
    const float bias = bias_C[n];
#pragma unroll
    for (int v = 0; v < 8; ++v) ytile[mb + v][n] = (__bf16)(acc[v] + bias);
  }
  __syncthreads();
  // Phase 2: y2 = y@Wp^T+bp ; g = sigmoid(x@Wg^T+bg) ; out = g*y2 + (1-g)*x
#pragma unroll 1
  for (int j = 0; j < 8; ++j) {
    const int n0 = (wave * 8 + j) * 16;
    v8f accY = {}, accG = {};
#pragma unroll 1
    for (int k = 0; k < H_; k += 32) {
      v16bf ay = frag_a_bf16(&ytile[0][k], 1032);
      v16bf by = frag_b_f32(Wp + (size_t)n0 * H_ + k, H_);
      accY = wmma_bf16(ay, by, accY);
      v16bf ax = frag_a_f32(x + (size_t)r0 * H_ + k, H_);
      v16bf bgf = frag_b_f32(Wg + (size_t)n0 * H_ + k, H_);
      accG = wmma_bf16(ax, bgf, accG);
    }
    const int n = n0 + (l & 15);
    const int mb = (l >> 4) * 8;
    const float bpv = bp[n], bgv = bg[n];
#pragma unroll
    for (int v = 0; v < 8; ++v) {
      const size_t idx = (size_t)(r0 + mb + v) * H_ + n;
      const float y2 = accY[v] + bpv;
      const float gv = accG[v] + bgv;
      const float sg = 1.0f / (1.0f + __expf(-gv));
      out[idx] = sg * y2 + (1.0f - sg) * x[idx];
    }
  }
}

extern "C" void kernel_launch(void* const* d_in, const int* in_sizes, int n_in,
                              void* d_out, int out_size, void* d_ws, size_t ws_size,
                              hipStream_t stream) {
  (void)in_sizes; (void)n_in; (void)out_size; (void)ws_size;
  const float* x     = (const float*)d_in[0];
  const float* W_in  = (const float*)d_in[1];
  const float* b_in  = (const float*)d_in[2];
  const float* A     = (const float*)d_in[3];
  const float* Bv    = (const float*)d_in[4];
  const float* C     = (const float*)d_in[5];
  const float* biasA = (const float*)d_in[6];
  const float* biasC = (const float*)d_in[7];
  const float* Wg    = (const float*)d_in[8];
  const float* bg    = (const float*)d_in[9];
  const float* Wp    = (const float*)d_in[10];
  const float* bp    = (const float*)d_in[11];
  float* out = (float*)d_out;

  uint8_t* ws = (uint8_t*)d_ws;
  unsigned int* cnt = (unsigned int*)ws;            // barrier counter (zeroed below)
  float* xs = (float*)(ws + 256);                   // x_state [B*T, S]  (32 MB)
  float* hs = xs + (size_t)B_ * T_ * S_;            // hs      [B, T, S] (32 MB)

  (void)hipMemsetAsync(d_ws, 0, 256, stream);       // graph-capture-safe reset
  k_inproj<<<dim3((B_ * T_) / 16), dim3(256), 0, stream>>>(x, W_in, b_in, xs);
  k_scan<<<dim3(S_ / 32), dim3(256), 0, stream>>>(xs, A, Bv, biasA, hs, cnt);
  k_out<<<dim3((B_ * T_) / 16), dim3(256), 0, stream>>>(hs, C, biasC, x, Wg, bg, Wp, bp, out);
}
